// SignaturedAsPoolingModel_67224828117656
// MI455X (gfx1250) — compile-verified, matrix operand order checked by reading, not run on hardware
//
#include <hip/hip_runtime.h>

typedef __attribute__((ext_vector_type(16))) _Float16 v16h;
typedef __attribute__((ext_vector_type(8)))  _Float16 v8h;
typedef __attribute__((ext_vector_type(8)))  float    v8f;
typedef __attribute__((ext_vector_type(4)))  unsigned int u32x4;
typedef __attribute__((ext_vector_type(8)))  int      i32x8;
typedef __attribute__((ext_vector_type(4)))  int      i32x4;

#define N_BATCH 2048
#define C_IN    64
#define C_SIG   32
#define FDIM    1056
#define NCLS    100
#define NP_LIN  1152   // lin_w rows padded to multiple of 128
#define NP_FC   128    // fc_w rows padded to multiple of 128

#if __has_builtin(__builtin_amdgcn_tensor_load_to_lds)
#define USE_TDM 1
#else
#define USE_TDM 0
#endif

// ---------------------------------------------------------------------------
// Kernel 1: conv(64->32, 3x3, pad1) + Zhang depth-2 signature for one image.
// ---------------------------------------------------------------------------
__global__ __launch_bounds__(256) void conv_sig_kernel(
    const float* __restrict__ feats, const float* __restrict__ conv_w,
    float* __restrict__ sig)
{
  const int n = blockIdx.x;
  const int t = threadIdx.x;
  __shared__ float xs[C_IN * 100];   // zero-padded 10x10 per input channel
  __shared__ float ys[C_SIG * 64];   // conv output 8x8 per sig channel
  __shared__ float Ds[C_SIG * 49];   // double increments 7x7
  __shared__ float Ps[C_SIG * 49];   // inclusive 2D prefix sums of D

  for (int idx = t; idx < C_IN * 100; idx += 256) xs[idx] = 0.f;
  __syncthreads();
  const float* fin = feats + (size_t)n * (C_IN * 64);
  for (int idx = t; idx < C_IN * 64; idx += 256) {
    int c = idx >> 6, p = idx & 63, i = p >> 3, j = p & 7;
    xs[c * 100 + (i + 1) * 10 + (j + 1)] = fin[idx];
  }
  __syncthreads();

  {
    const int px = t & 63, ocb = t >> 6;
    const int i = px >> 3, j = px & 7;
    float acc[8];
    #pragma unroll
    for (int u = 0; u < 8; ++u) acc[u] = 0.f;
    for (int ic = 0; ic < C_IN; ++ic) {
      const int xb = ic * 100 + i * 10 + j;
      #pragma unroll
      for (int kh = 0; kh < 3; ++kh) {
        #pragma unroll
        for (int kw = 0; kw < 3; ++kw) {
          const float xv = xs[xb + kh * 10 + kw];
          const int  wb = ic * 9 + kh * 3 + kw;
          #pragma unroll
          for (int u = 0; u < 8; ++u)
            acc[u] = fmaf(conv_w[(ocb * 8 + u) * (C_IN * 9) + wb], xv, acc[u]);
        }
      }
    }
    #pragma unroll
    for (int u = 0; u < 8; ++u) ys[(ocb * 8 + u) * 64 + px] = acc[u];
  }
  __syncthreads();

  for (int idx = t; idx < C_SIG * 49; idx += 256) {
    int c = idx / 49, rem = idx % 49, i = rem / 7, j = rem % 7;
    Ds[idx] = ys[c * 64 + (i + 1) * 8 + (j + 1)] - ys[c * 64 + (i + 1) * 8 + j]
            - ys[c * 64 + i * 8 + (j + 1)]       + ys[c * 64 + i * 8 + j];
  }
  __syncthreads();

  if (t < C_SIG) {
    const int c = t;
    float s = 0.f;
    for (int k = 0; k < 49; ++k) s += Ds[c * 49 + k];
    sig[(size_t)n * FDIM + c] = s;
    for (int i = 0; i < 7; ++i) {
      float run = 0.f;
      for (int j = 0; j < 7; ++j) {
        run += Ds[c * 49 + i * 7 + j];
        Ps[c * 49 + i * 7 + j] = run + (i ? Ps[c * 49 + (i - 1) * 7 + j] : 0.f);
      }
    }
  }
  __syncthreads();

  for (int idx = t; idx < C_SIG * C_SIG; idx += 256) {
    int c = idx >> 5, c2 = idx & 31;
    float s = 0.f;
    #pragma unroll
    for (int ii = 0; ii < 6; ++ii)
      #pragma unroll
      for (int jj = 0; jj < 6; ++jj)
        s = fmaf(Ps[c * 49 + ii * 7 + jj], Ds[c2 * 49 + (ii + 1) * 7 + (jj + 1)], s);
    sig[(size_t)n * FDIM + C_SIG + idx] = s;
  }
}

// ---------------------------------------------------------------------------
// Kernel 2: BatchNorm batch statistics -> per-feature scale/shift
// ---------------------------------------------------------------------------
__global__ __launch_bounds__(256) void bn_stats_kernel(
    const float* __restrict__ sig, const float* __restrict__ gamma,
    const float* __restrict__ beta, float* __restrict__ scale,
    float* __restrict__ shift)
{
  const int f = blockIdx.x;
  const int t = threadIdx.x;
  float s = 0.f, s2 = 0.f;
  for (int nidx = t; nidx < N_BATCH; nidx += 256) {
    float v = sig[(size_t)nidx * FDIM + f];
    s += v; s2 += v * v;
  }
  __shared__ float rs[256], rq[256];
  rs[t] = s; rq[t] = s2;
  __syncthreads();
  for (int off = 128; off > 0; off >>= 1) {
    if (t < off) { rs[t] += rs[t + off]; rq[t] += rq[t + off]; }
    __syncthreads();
  }
  if (t == 0) {
    float mean = rs[0] * (1.f / N_BATCH);
    float var  = rq[0] * (1.f / N_BATCH) - mean * mean;
    float sc   = gamma[f] * rsqrtf(var + 1e-5f);
    scale[f] = sc;
    shift[f] = beta[f] - mean * sc;
  }
}

// Kernel 3: normalize + convert to f16 activations
__global__ __launch_bounds__(256) void norm_pack_kernel(
    const float* __restrict__ sig, const float* __restrict__ scale,
    const float* __restrict__ shift, _Float16* __restrict__ hn)
{
  int idx = blockIdx.x * 256 + threadIdx.x;
  if (idx >= N_BATCH * FDIM) return;
  int f = idx % FDIM;
  hn[idx] = (_Float16)(sig[idx] * scale[f] + shift[f]);
}

// Kernel 4: pack a (rows_valid x FDIM) f32 matrix to f16, zero-padded rows
__global__ __launch_bounds__(256) void pack_pad_kernel(
    const float* __restrict__ w, _Float16* __restrict__ wh,
    int rows_valid, int rows_total)
{
  int idx = blockIdx.x * 256 + threadIdx.x;
  if (idx >= rows_total * FDIM) return;
  int o = idx / FDIM, k = idx % FDIM;
  wh[idx] = (o < rows_valid) ? (_Float16)w[o * FDIM + k] : (_Float16)0.f;
}

// ---------------------------------------------------------------------------
// TDM: issue a 2-D tile load (rows x cols f16 elements, row stride =
// stride_elems) from global into LDS.  Descriptor packing per CDNA5 ISA
// CDNA5 ISA (group0: count/lds_addr/global_addr/type=2;
// group1: data_size=2B, tensor dims == tile dims, dim0 stride).
// This toolchain exposes the 6-arg builtin:
//   (u32x4 g0, i32x8 g1, i32x4 g2, i32x4 g3, i32x8 pad, i32 cpol)
// ---------------------------------------------------------------------------
#if USE_TDM
__device__ __forceinline__ void tdm_load_tile_2d(
    const _Float16* gptr, const _Float16* lptr,
    unsigned rows, unsigned cols, unsigned stride_elems)
{
  unsigned long long ga = (unsigned long long)gptr;
  unsigned lds_off = (unsigned)(unsigned long long)lptr;
  u32x4 g0;
  g0.x = 1u;                                            // count=1 user D#
  g0.y = lds_off;                                       // lds_addr
  g0.z = (unsigned)ga;                                  // global_addr[31:0]
  g0.w = (unsigned)((ga >> 32) & 0x01ffffffull) | (2u << 30); // addr[56:32]|type=2
  unsigned long long s0 = (unsigned long long)stride_elems;
  i32x8 g1;
  g1[0] = (int)(1u << 16);                              // data_size code 1 (2B)
  g1[1] = (int)((cols & 0xffffu) << 16);                // tensor_dim0[15:0]
  g1[2] = (int)((cols >> 16) | ((rows & 0xffffu) << 16)); // td0 hi | td1 lo
  g1[3] = (int)((rows >> 16) | ((cols & 0xffffu) << 16)); // td1 hi | tile_dim0
  g1[4] = (int)(rows & 0xffffu);                        // tile_dim1 (tile_dim2=0)
  g1[5] = (int)(unsigned)(s0 & 0xffffffffull);          // dim0_stride lo
  g1[6] = (int)(unsigned)((s0 >> 32) & 0xffffull);      // dim0_stride hi
  g1[7] = 0;
  i32x4 z4 = {0, 0, 0, 0};
  i32x8 z8 = {0, 0, 0, 0, 0, 0, 0, 0};
  __builtin_amdgcn_tensor_load_to_lds(g0, g1, z4, z4, z8, 0);
}
#endif

// fallback cooperative stage (used if TDM builtin is absent)
__device__ __forceinline__ void lds_stage_tile(
    const _Float16* __restrict__ g, _Float16* __restrict__ l,
    int stride_elems, int t)
{
  // 128 rows x 32 cols, 8-element chunks: 512 chunks / 256 threads
  for (int idx = t; idx < 128 * 4; idx += 256) {
    int row = idx >> 2, c = (idx & 3) * 8;
    *(v8h*)(l + row * 32 + c) = *(const v8h*)(g + (size_t)row * stride_elems + c);
  }
}

// ---------------------------------------------------------------------------
// WMMA GEMM with LDS tiling + TDM double-buffered staging.
//   C[M x nvalid] = A(M x K) * B(Nt x K)^T + bias, f16 in / f32 accum.
// Block = 128x128 output tile, 8 waves in a 2(M) x 4(N) grid, each wave owns
// a 64x32 sub-tile = 4x2 16x16 WMMA accumulators.  grid = (Nt/128, M/128).
// ---------------------------------------------------------------------------
__global__ __launch_bounds__(256) void gemm_wmma_tdm_kernel(
    const _Float16* __restrict__ A, const _Float16* __restrict__ B,
    const float* __restrict__ bias, _Float16* __restrict__ Ch,
    float* __restrict__ Cf, int K, int ldc, int nvalid)
{
  __shared__ _Float16 Asm[2][128 * 32];   // 8 KB x2
  __shared__ _Float16 Bsm[2][128 * 32];   // 8 KB x2

  const int wave = threadIdx.x >> 5;
  const int lane = threadIdx.x & 31;
  const int half = lane >> 4;
  const int r    = lane & 15;
  const int wm   = (wave >> 2) * 64;    // wave M offset inside block tile
  const int wn   = (wave & 3) * 32;     // wave N offset inside block tile
  const int mblk = blockIdx.y * 128;
  const int nblk = blockIdx.x * 128;

  union V16 { v16h v; v8h h[2]; };
  v8f acc[4][2];
  #pragma unroll
  for (int mi = 0; mi < 4; ++mi)
    #pragma unroll
    for (int ni = 0; ni < 2; ++ni) acc[mi][ni] = (v8f){};

  const int nsteps = K / 32;

  // prologue: stage k-step 0 into buffer 0
#if USE_TDM
  if (wave == 0) {
    tdm_load_tile_2d(A + (size_t)mblk * K, &Asm[0][0], 128, 32, K);
    tdm_load_tile_2d(B + (size_t)nblk * K, &Bsm[0][0], 128, 32, K);
    __builtin_amdgcn_s_wait_tensorcnt(0);
  }
#else
  lds_stage_tile(A + (size_t)mblk * K, &Asm[0][0], K, threadIdx.x);
  lds_stage_tile(B + (size_t)nblk * K, &Bsm[0][0], K, threadIdx.x);
#endif
  __syncthreads();

  for (int ks = 0; ks < nsteps; ++ks) {
    const int cur = ks & 1;
    // kick off next tile into the other buffer while we compute
    if (ks + 1 < nsteps) {
      const int k0n = (ks + 1) * 32;
#if USE_TDM
      if (wave == 0) {
        tdm_load_tile_2d(A + (size_t)mblk * K + k0n, &Asm[cur ^ 1][0], 128, 32, K);
        tdm_load_tile_2d(B + (size_t)nblk * K + k0n, &Bsm[cur ^ 1][0], 128, 32, K);
      }
#else
      lds_stage_tile(A + (size_t)mblk * K + k0n, &Asm[cur ^ 1][0], K, threadIdx.x);
      lds_stage_tile(B + (size_t)nblk * K + k0n, &Bsm[cur ^ 1][0], K, threadIdx.x);
#endif
    }

    // fragments from LDS (layout per ISA 7.12.2: lane -> row lane%16,
    // K chunks {half*8..+7} and {16+half*8..+7})
    V16 af[4], bf[2];
    #pragma unroll
    for (int mi = 0; mi < 4; ++mi) {
      const _Float16* p = &Asm[cur][(wm + mi * 16 + r) * 32 + half * 8];
      af[mi].h[0] = *(const v8h*)(p);
      af[mi].h[1] = *(const v8h*)(p + 16);
    }
    #pragma unroll
    for (int ni = 0; ni < 2; ++ni) {
      const _Float16* p = &Bsm[cur][(wn + ni * 16 + r) * 32 + half * 8];
      bf[ni].h[0] = *(const v8h*)(p);
      bf[ni].h[1] = *(const v8h*)(p + 16);
    }
    #pragma unroll
    for (int mi = 0; mi < 4; ++mi)
      #pragma unroll
      for (int ni = 0; ni < 2; ++ni)
        acc[mi][ni] = __builtin_amdgcn_wmma_f32_16x16x32_f16(
            false, af[mi].v, false, bf[ni].v, (short)0, acc[mi][ni], false, false);

    if (ks + 1 < nsteps) {
#if USE_TDM
      if (wave == 0) __builtin_amdgcn_s_wait_tensorcnt(0);
#endif
      __syncthreads();
    }
  }

  // epilogue: D layout -> lane holds col lane%16, rows e + 8*half
  #pragma unroll
  for (int ni = 0; ni < 2; ++ni) {
    const int col = nblk + wn + ni * 16 + r;
    if (col >= nvalid) continue;
    const float bv = bias[col];
    #pragma unroll
    for (int mi = 0; mi < 4; ++mi) {
      #pragma unroll
      for (int e = 0; e < 8; ++e) {
        const int row = mblk + wm + mi * 16 + e + 8 * half;
        const float v = acc[mi][ni][e] + bv;
        if (Cf) Cf[(size_t)row * ldc + col] = v;
        else    Ch[(size_t)row * ldc + col] = (_Float16)v;
      }
    }
  }
}

// ---------------------------------------------------------------------------
extern "C" void kernel_launch(void* const* d_in, const int* in_sizes, int n_in,
                              void* d_out, int out_size, void* d_ws, size_t ws_size,
                              hipStream_t stream) {
  const float* feats  = (const float*)d_in[0];
  const float* conv_w = (const float*)d_in[1];
  const float* gamma  = (const float*)d_in[2];
  const float* beta   = (const float*)d_in[3];
  const float* lin_w  = (const float*)d_in[4];
  const float* lin_b  = (const float*)d_in[5];
  const float* fc_w   = (const float*)d_in[6];
  const float* fc_b   = (const float*)d_in[7];
  float* out = (float*)d_out;

  char* ws = (char*)d_ws;
  size_t off = 0;
  auto take = [&](size_t bytes) { char* p = ws + off; off = (off + bytes + 255) & ~(size_t)255; return p; };
  float*    sig   = (float*)   take((size_t)N_BATCH * FDIM * 4);
  float*    scale = (float*)   take(FDIM * 4);
  float*    shift = (float*)   take(FDIM * 4);
  _Float16* hn    = (_Float16*)take((size_t)N_BATCH * FDIM * 2);
  _Float16* lwh   = (_Float16*)take((size_t)NP_LIN * FDIM * 2);
  _Float16* fwh   = (_Float16*)take((size_t)NP_FC * FDIM * 2);
  _Float16* h2h   = (_Float16*)take((size_t)N_BATCH * FDIM * 2);
  (void)ws_size; (void)in_sizes; (void)n_in; (void)out_size;

  // 1) conv + signature
  conv_sig_kernel<<<N_BATCH, 256, 0, stream>>>(feats, conv_w, sig);

  // 2) batchnorm statistics
  bn_stats_kernel<<<FDIM, 256, 0, stream>>>(sig, gamma, beta, scale, shift);

  // 3) normalize + f16 pack
  {
    int cnt = N_BATCH * FDIM;
    norm_pack_kernel<<<(cnt + 255) / 256, 256, 0, stream>>>(sig, scale, shift, hn);
  }
  // 4) weight packs (zero-padded N)
  {
    int cnt = NP_LIN * FDIM;
    pack_pad_kernel<<<(cnt + 255) / 256, 256, 0, stream>>>(lin_w, lwh, FDIM, NP_LIN);
    int cnt2 = NP_FC * FDIM;
    pack_pad_kernel<<<(cnt2 + 255) / 256, 256, 0, stream>>>(fc_w, fwh, NCLS, NP_FC);
  }

  // 5) GEMM1: h2 = hn @ lin_w^T + lin_b  -> f16 (2048 x 1056, K=1056)
  {
    dim3 grid(NP_LIN / 128, N_BATCH / 128);
    gemm_wmma_tdm_kernel<<<grid, 256, 0, stream>>>(hn, lwh, lin_b, h2h, nullptr,
                                                   FDIM, FDIM, FDIM);
  }
  // 6) GEMM2: out = h2 @ fc_w^T + fc_b  -> f32 d_out (2048 x 100, K=1056)
  {
    dim3 grid(NP_FC / 128, N_BATCH / 128);
    gemm_wmma_tdm_kernel<<<grid, 256, 0, stream>>>(h2h, fwh, fc_b, nullptr, out,
                                                   FDIM, NCLS, NCLS);
  }
}
